// AdaptiveRankSemiseparableLayer_72258529788076
// MI455X (gfx1250) — matrix operand order repaired and checked by reading, not compiled, not run on hardware
//
#include <hip/hip_runtime.h>
#include <hip/hip_bf16.h>

// Problem constants (from reference): B=4, L=4096, D=1024, R=32, GH=256
#define BB  4
#define LL  4096
#define DD  1024
#define RR  32
#define GHH 256
#define MTOT (BB * LL)   // 16384 rows

typedef __attribute__((ext_vector_type(16))) __bf16 v16bf;
typedef __attribute__((ext_vector_type(8)))  __bf16 v8bf;
typedef __attribute__((ext_vector_type(8)))  float  v8f;
typedef __attribute__((ext_vector_type(4)))  unsigned int u32x4;
typedef __attribute__((ext_vector_type(4)))  int i32x4;
typedef __attribute__((ext_vector_type(8)))  int i32x8;

// ---- helpers --------------------------------------------------------------

// f32 -> bf16 round-to-nearest-even (used only in one-time prep + epilogues)
__device__ __forceinline__ __bf16 f2bf(float f) {
    union { float f; unsigned u; } v; v.f = f;
    unsigned r = (v.u + 0x7FFFu + ((v.u >> 16) & 1u)) >> 16;
    unsigned short h = (unsigned short)r;
    __bf16 b;
    __builtin_memcpy(&b, &h, 2);
    return b;
}

__device__ __forceinline__ v8f wmma_bf16(v16bf a, v16bf b, v8f c) {
    return __builtin_amdgcn_wmma_f32_16x16x32_bf16(
        false, a, false, b, (short)0, c, false, false);
}

// A-fragment from a row-major bf16 tile (LDS or global), per ISA 16-bit A 16x32:
// lanes 0-15: M=lane, K {k0..k0+7, k0+16..k0+23}; lanes 16-31: M=lane-16,
// K {k0+8..k0+15, k0+24..k0+31}. Two 16B vector loads, no conversion.
__device__ __forceinline__ v16bf loadA_bf16(const __bf16* s, int ld, int k0, int lane) {
    int m  = lane & 15;
    int kb = k0 + ((lane >> 4) << 3);
    const __bf16* p = s + (size_t)m * ld + kb;
    v8bf lo = *(const v8bf*)p;
    v8bf hi = *(const v8bf*)(p + 16);
    return __builtin_shufflevector(lo, hi, 0, 1, 2, 3, 4, 5, 6, 7,
                                   8, 9, 10, 11, 12, 13, 14, 15);
}

// B-fragment from the pre-packed fragment buffer: one aligned 32-byte load.
__device__ __forceinline__ v16bf loadB_frag(const __bf16* F, int NT, int kt, int nt, int lane) {
    return ((const v16bf*)F)[(size_t)(kt * NT + nt) * 32 + lane];
}

// ---- prep kernels (run once, off the critical loops) -----------------------

__global__ __launch_bounds__(256) void prep_xbf(const float* __restrict__ x,
                                                __bf16* __restrict__ xb, int n) {
    int i = blockIdx.x * blockDim.x + threadIdx.x;
    if (i < n) xb[i] = f2bf(x[i]);
}

// Pack f32 weight W[K][N] (row-major, ld=N) into bf16 WMMA B-fragment layout:
// dst[((kt*NT+nt)*32 + lane)*16 + e] = W[kt*32 + (lane>>4)*8 + (e<8 ? e : e+8)][nt*16 + (lane&15)]
__global__ __launch_bounds__(128) void prep_bfrag(const float* __restrict__ W,
                                                  int K, int N,
                                                  __bf16* __restrict__ dst) {
    int t = blockIdx.x * blockDim.x + threadIdx.x;   // = tile*32 + lane
    int NT = N / 16, KT = K / 32;
    if (t >= KT * NT * 32) return;
    int lane = t & 31;
    int tile = t >> 5;
    int nt = tile % NT, kt = tile / NT;
    int n  = nt * 16 + (lane & 15);
    int kb = kt * 32 + ((lane >> 4) << 3);
    __bf16* d = dst + (size_t)t * 16;
#pragma unroll
    for (int e = 0; e < 16; ++e) d[e] = f2bf(W[(size_t)(kb + (e < 8 ? e : e + 8)) * N + n]);
}

// ---- Kernel 1: fused x->(h,relu)->gate, U, V ; store gate*U and V ----------
// One wave32 per block, one 16-row tile. x tile staged into LDS via TDM.
__global__ __launch_bounds__(32) void ar_front_kernel(
    const __bf16* __restrict__ xb,
    const __bf16* __restrict__ gw1f, const float* __restrict__ gb1,
    const __bf16* __restrict__ gw2f, const float* __restrict__ gb2,
    const __bf16* __restrict__ uwf,  const __bf16* __restrict__ vwf,
    float* __restrict__ gU, float* __restrict__ Vout)
{
    __shared__ __bf16 xs[16 * DD];    // 32 KB: 16-row x tile (bf16)
    __shared__ __bf16 hs[16 * GHH];   //  8 KB: relu(h) tile (bf16)

    const int lane = threadIdx.x;
    const int row0 = blockIdx.x * 16;
    const __bf16* xrow = xb + (size_t)row0 * DD;

    if (lane == 0) {
        __builtin_prefetch(gw1f, 0, 1);   // global_prefetch_b8: L2-resident weights
        __builtin_prefetch(gw2f, 0, 1);
        __builtin_prefetch(uwf, 0, 1);
        __builtin_prefetch(vwf, 0, 1);
    }

#if __has_builtin(__builtin_amdgcn_tensor_load_to_lds)
    {
        // Tensor DMA descriptor (ISA 08_async_tensor §8): 2D bf16 tensor
        // (dim0 = D = 1024, dim1 = M = 16384), tile 1024 x 16 at row0.
        unsigned long long ga = (unsigned long long)xrow;
        unsigned ldsa = (unsigned)(size_t)(&xs[0]);
        u32x4 g0;
        g0.x = 1u;                                     // count=1 (valid D#)
        g0.y = ldsa;                                   // lds_addr (bytes)
        g0.z = (unsigned)(ga & 0xFFFFFFFFu);           // global_addr[31:0]
        g0.w = (unsigned)((ga >> 32) & 0x01FFFFFFu)    // global_addr[56:32]
             | (2u << 30);                             // type = 2 ("image")
        i32x8 g1;
        g1[0] = (int)(1u << 16);                       // data_size = 1 -> 2 bytes
        g1[1] = (int)((unsigned)DD << 16);             // tensor_dim0[15:0] = 1024
        g1[2] = (int)((unsigned)MTOT << 16);           // tensor_dim1[15:0] = 16384
        g1[3] = (int)((unsigned)DD << 16);             // tile_dim0 = 1024
        g1[4] = 16;                                    // tile_dim1 = 16 rows
        g1[5] = DD;                                    // tensor_dim0_stride = 1024
        g1[6] = 0;
        g1[7] = 0;
        i32x4 gz = {0, 0, 0, 0};
#if __clang_major__ >= 23
        i32x8 gz8 = {0, 0, 0, 0, 0, 0, 0, 0};
        __builtin_amdgcn_tensor_load_to_lds(g0, g1, gz, gz, gz8, 0);
#else
        __builtin_amdgcn_tensor_load_to_lds(g0, g1, gz, gz, 0);
#endif
#if __has_builtin(__builtin_amdgcn_s_wait_tensorcnt)
        __builtin_amdgcn_s_wait_tensorcnt(0);
#else
        asm volatile("s_wait_tensorcnt 0x0" ::: "memory");
#endif
    }
#else
    // Fallback: cooperative wide-vector copy global(bf16) -> LDS
    for (int i = lane * 8; i < 16 * DD; i += 32 * 8)
        *(v8bf*)(xs + i) = *(const v8bf*)(xrow + i);
#endif
    __syncthreads();

    const int nloc = lane & 15;
    const int mb   = (lane >> 4) * 8;

    // ---- h = relu(x @ gw1 + gb1) -> LDS (bf16, A-ready row-major)
    for (int nt = 0; nt < GHH / 16; ++nt) {
        v8f acc = {};
        for (int kt = 0; kt < DD / 32; ++kt) {
            v16bf a = loadA_bf16(xs, DD, kt * 32, lane);
            v16bf b = loadB_frag(gw1f, GHH / 16, kt, nt, lane);
            acc = wmma_bf16(a, b, acc);
        }
        float bias = gb1[nt * 16 + nloc];
#pragma unroll
        for (int v = 0; v < 8; ++v) {
            float h = acc[v] + bias;
            hs[(mb + v) * GHH + nt * 16 + nloc] = f2bf(h > 0.f ? h : 0.f);
        }
    }
    __syncthreads();

    // ---- U = x @ Uw ; V = x @ Vw  (N = 32 -> 2 tiles each)
    v8f uacc[2] = {};
    v8f vacc[2] = {};
    for (int kt = 0; kt < DD / 32; ++kt) {
        v16bf a = loadA_bf16(xs, DD, kt * 32, lane);
#pragma unroll
        for (int nt = 0; nt < 2; ++nt) {
            uacc[nt] = wmma_bf16(a, loadB_frag(uwf, 2, kt, nt, lane), uacc[nt]);
            vacc[nt] = wmma_bf16(a, loadB_frag(vwf, 2, kt, nt, lane), vacc[nt]);
        }
    }

    // ---- gate = sigmoid(h @ gw2 + gb2)
    v8f gacc[2] = {};
    for (int kt = 0; kt < GHH / 32; ++kt) {
        v16bf a = loadA_bf16(hs, GHH, kt * 32, lane);
#pragma unroll
        for (int nt = 0; nt < 2; ++nt)
            gacc[nt] = wmma_bf16(a, loadB_frag(gw2f, 2, kt, nt, lane), gacc[nt]);
    }

    // ---- combine: store gate*U and V (C layout: n = lane&15, m = mb+v)
#pragma unroll
    for (int nt = 0; nt < 2; ++nt) {
        float bias = gb2[nt * 16 + nloc];
#pragma unroll
        for (int v = 0; v < 8; ++v) {
            int r = nt * 16 + nloc;
            int m = mb + v;
            float g = 1.f / (1.f + __expf(-(gacc[nt][v] + bias)));
            size_t idx = (size_t)(row0 + m) * RR + r;
            gU[idx]   = g * uacc[nt][v];
            Vout[idx] = vacc[nt][v];
        }
    }
}

// ---- Kernel 2: causal cumsum over L per (b,r); glob = gateU * cumsum(V) ----
// Emits glob directly in bf16 (A-operand-ready for kernel 3).
__global__ __launch_bounds__(128) void ar_scan_kernel(
    const float* __restrict__ gU, const float* __restrict__ Vin,
    __bf16* __restrict__ globbf)
{
    const int b = blockIdx.x / RR;
    const int r = blockIdx.x % RR;
    const int t = threadIdx.x;
    const size_t base = (size_t)b * LL * RR + r;

    const int l0 = t * 32;
    float vals[32];
    float run = 0.f;
#pragma unroll
    for (int i = 0; i < 32; ++i) {
        run += Vin[base + (size_t)(l0 + i) * RR];
        vals[i] = run;
    }

    __shared__ float sums[128];
    sums[t] = run;
    __syncthreads();
    for (int off = 1; off < 128; off <<= 1) {
        float v = (t >= off) ? sums[t - off] : 0.f;
        __syncthreads();
        sums[t] += v;
        __syncthreads();
    }
    float carry = sums[t] - run;

#pragma unroll
    for (int i = 0; i < 32; ++i) {
        size_t idx = base + (size_t)(l0 + i) * RR;
        globbf[idx] = f2bf(gU[idx] * (vals[i] + carry));
    }
}

// ---- Kernel 3: y = glob @ out_w + depthwise_conv3(x) + out_b ---------------
// 4 waves/block; wave handles 16 rows x 256 cols (16 N-tiles, K=32 -> 1 WMMA).
__global__ __launch_bounds__(128) void ar_out_kernel(
    const __bf16* __restrict__ globbf,
    const __bf16* __restrict__ owf, const float* __restrict__ out_b,
    const float* __restrict__ x,    const float* __restrict__ conv_w,
    float* __restrict__ out)
{
    const int lane = threadIdx.x & 31;
    const int wave = threadIdx.x >> 5;
    const int row0 = blockIdx.x * 16;
    const int ncol0 = wave * 256;

    if (threadIdx.x == 0) __builtin_prefetch(owf, 0, 1);

    // A fragment: glob tile 16 x 32 (whole K), straight bf16 loads
    v16bf a = loadA_bf16(globbf + (size_t)row0 * RR, RR, 0, lane);

    const int nloc = lane & 15;
    const int mb   = (lane >> 4) * 8;

    for (int nt = 0; nt < 16; ++nt) {
        const int n0 = ncol0 + nt * 16;
        v16bf b = loadB_frag(owf, DD / 16, 0, n0 / 16, lane);
        v8f c = {};
        c = wmma_bf16(a, b, c);

        const int n = n0 + nloc;
        const float bias = out_b[n];
        const float w0 = conv_w[n * 3 + 0];
        const float w1 = conv_w[n * 3 + 1];
        const float w2 = conv_w[n * 3 + 2];
#pragma unroll
        for (int v = 0; v < 8; ++v) {
            const int row = row0 + mb + v;
            const int l   = row & (LL - 1);
            const float* xr = x + (size_t)row * DD + n;
            float conv = w1 * xr[0];
            if (l > 0)      conv += w0 * xr[-DD];
            if (l < LL - 1) conv += w2 * xr[DD];
            out[(size_t)row * DD + n] = c[v] + conv + bias;
        }
    }
}

// ---- host-side launcher ----------------------------------------------------
extern "C" void kernel_launch(void* const* d_in, const int* in_sizes, int n_in,
                              void* d_out, int out_size, void* d_ws, size_t ws_size,
                              hipStream_t stream) {
    (void)in_sizes; (void)n_in; (void)out_size; (void)ws_size;

    const float* x      = (const float*)d_in[0];
    const float* gw1    = (const float*)d_in[1];
    const float* gb1    = (const float*)d_in[2];
    const float* gw2    = (const float*)d_in[3];
    const float* gb2    = (const float*)d_in[4];
    const float* Uw     = (const float*)d_in[5];
    const float* Vw     = (const float*)d_in[6];
    const float* conv_w = (const float*)d_in[7];
    const float* out_w  = (const float*)d_in[8];
    const float* out_b  = (const float*)d_in[9];
    float* out = (float*)d_out;

    // Workspace partition (256B aligned slabs)
    const size_t xbN   = (size_t)MTOT * DD;                          // bf16
    const size_t gw1fN = (size_t)(DD / 32) * (GHH / 16) * 32 * 16;   // bf16
    const size_t gw2fN = (size_t)(GHH / 32) * (RR / 16) * 32 * 16;
    const size_t uwfN  = (size_t)(DD / 32) * (RR / 16) * 32 * 16;
    const size_t owfN  = (size_t)(RR / 32) * (DD / 16) * 32 * 16;
    const size_t gUN   = (size_t)MTOT * RR;                          // f32
    auto al = [](size_t v) { return (v + 255) & ~(size_t)255; };

    char* base = (char*)d_ws;
    size_t off = 0;
    __bf16* xb     = (__bf16*)(base + off); off = al(off + xbN * 2);
    __bf16* gw1f   = (__bf16*)(base + off); off = al(off + gw1fN * 2);
    __bf16* gw2f   = (__bf16*)(base + off); off = al(off + gw2fN * 2);
    __bf16* uwf    = (__bf16*)(base + off); off = al(off + uwfN * 2);
    __bf16* vwf    = (__bf16*)(base + off); off = al(off + uwfN * 2);
    __bf16* owf    = (__bf16*)(base + off); off = al(off + owfN * 2);
    float*  gU     = (float*)(base + off);  off = al(off + gUN * 4);
    float*  Vtmp   = (float*)(base + off);  off = al(off + gUN * 4);
    __bf16* globbf = (__bf16*)(base + off); off = al(off + gUN * 2);

    // One-time conversions / fragment packing
    prep_xbf<<<(int)((xbN + 255) / 256), 256, 0, stream>>>(x, xb, (int)xbN);
    prep_bfrag<<<(int)((gw1fN / 16 + 127) / 128), 128, 0, stream>>>(gw1, DD, GHH, gw1f);
    prep_bfrag<<<(int)((gw2fN / 16 + 127) / 128), 128, 0, stream>>>(gw2, GHH, RR, gw2f);
    prep_bfrag<<<(int)((uwfN / 16 + 127) / 128), 128, 0, stream>>>(Uw, DD, RR, uwf);
    prep_bfrag<<<(int)((uwfN / 16 + 127) / 128), 128, 0, stream>>>(Vw, DD, RR, vwf);
    prep_bfrag<<<(int)((owfN / 16 + 127) / 128), 128, 0, stream>>>(out_w, RR, DD, owf);

    // Main pipeline
    ar_front_kernel<<<MTOT / 16, 32, 0, stream>>>(xb, gw1f, gb1, gw2f, gb2,
                                                  uwf, vwf, gU, Vtmp);
    ar_scan_kernel<<<BB * RR, 128, 0, stream>>>(gU, Vtmp, globbf);
    ar_out_kernel<<<MTOT / 16, 128, 0, stream>>>(globbf, owf, out_b, x, conv_w, out);
}
